// MultiHeadAttention_61856118997470
// MI455X (gfx1250) — matrix operand order repaired and checked by reading, not compiled, not run on hardware
//
#include <hip/hip_runtime.h>
#include <hip/hip_bf16.h>

// Problem constants (from reference)
#define DIMC  256                 // channel dim
#define HEADS 4
#define NBATCH 8
#define NSP   4096                // H*W = 64*64
#define OQKV  (3 * DIMC * HEADS)  // 3072
#define CO    (DIMC * HEADS)      // 1024
#define KSTEP 32
#define TM    128
#define TN    128

typedef __attribute__((ext_vector_type(16))) __bf16 v16bf;
typedef __attribute__((ext_vector_type(8)))  __bf16 v8bf;
typedef __attribute__((ext_vector_type(8)))  float  v8f;
typedef __attribute__((ext_vector_type(4)))  unsigned int u32x4;
typedef __attribute__((ext_vector_type(8)))  int i32x8;
typedef __attribute__((ext_vector_type(4)))  int i32x4;

__device__ __forceinline__ __bf16 f2bf(float f) {
  return static_cast<__bf16>(f);   // lowers to v_cvt_pk_bf16_f32 on gfx1250
}

// 32-byte async copy global -> LDS (per thread), tracked by ASYNCcnt.
// INST_OFFSET is added to BOTH lds and global addresses (ISA 15.18.3 op 98),
// so one address pair covers both 16B halves.
__device__ __forceinline__ void async_cp32(__bf16* lds, const __bf16* g) {
  unsigned ldsOff = (unsigned)(size_t)lds;   // low 32 bits of flat shared ptr = LDS offset
  asm volatile(
      "global_load_async_to_lds_b128 %0, %1, off\n\t"
      "global_load_async_to_lds_b128 %0, %1, off offset:16"
      :
      : "v"(ldsOff), "v"(g)
      : "memory");
}

__device__ __forceinline__ void wait_async0() {
  asm volatile("s_wait_asynccnt 0x0" ::: "memory");
}

// Tensor Data Mover: 2D tile (tileW elems x tileH rows, 2B elems) from global
// (row stride = rowStrideElems) into LDS as packed rows. D# layout per
// CDNA5 ISA. Issue from ONE wave per block.
// This toolchain's builtin takes 6 args (g0, g1, g2, g3, g4, cpol).
__device__ __forceinline__ void tdm_load_2d(unsigned ldsOff, const void* gaddr,
                                            unsigned rowStrideElems,
                                            unsigned tileW, unsigned tileH) {
  const unsigned long long ga = (unsigned long long)(size_t)gaddr;
  const unsigned d0s = rowStrideElems;   // tensor_dim0 (elements)
  const unsigned d1s = 32768u;           // tensor_dim1 (generous, in-bounds tiles)
  u32x4 g0;
  g0[0] = 1u;                                                   // count=1, user D#
  g0[1] = ldsOff;                                               // lds_addr (bytes)
  g0[2] = (unsigned)ga;                                         // global_addr[31:0]
  g0[3] = ((unsigned)(ga >> 32) & 0x01FFFFFFu) | 0x80000000u;   // ga[56:32] | type=2
  i32x8 g1;
  g1[0] = (int)(1u << 16);                                      // data_size = 2B
  g1[1] = (int)((d0s & 0xFFFFu) << 16);                         // tensor_dim0 lo
  g1[2] = (int)(((d0s >> 16) & 0xFFFFu) | ((d1s & 0xFFFFu) << 16)); // dim0 hi | dim1 lo
  g1[3] = (int)(((d1s >> 16) & 0xFFFFu) | ((tileW & 0xFFFFu) << 16)); // dim1 hi | tile_dim0
  g1[4] = (int)(tileH & 0xFFFFu);                               // tile_dim1 (tile_dim2=0)
  g1[5] = (int)rowStrideElems;                                  // tensor_dim0_stride lo
  g1[6] = 0;                                                    // stride hi | dim1_stride lo
  g1[7] = 0;
  const i32x4 z4 = {0, 0, 0, 0};                                // 2D: groups 2/3 unused
  const i32x8 z8 = {0, 0, 0, 0, 0, 0, 0, 0};
  __builtin_amdgcn_tensor_load_to_lds(g0, g1, z4, z4, z8, 0);
}

// ---------------- LDS tile fills: 256 threads fill a 128 x 32 bf16 tile ----
// Tile stored row-major [128][KSTEP]; "rows" are the fragment-lane axis.

__device__ __forceinline__ void fill_direct_f32(__bf16* dst, const float* src,
                                                int ld, int r0, int k0, int t) {
  const int row = t >> 1, col = (t & 1) * 16;
  const float4* g = (const float4*)(src + (size_t)(r0 + row) * ld + (k0 + col));
  float4 f[4]; f[0] = g[0]; f[1] = g[1]; f[2] = g[2]; f[3] = g[3];
  const float* fs = (const float*)f;
  __bf16* d = dst + row * KSTEP + col;
#pragma unroll
  for (int j = 0; j < 16; ++j) d[j] = f2bf(fs[j]);
}

__device__ __forceinline__ void fill_trans_f32(__bf16* dst, const float* src,
                                               int ld, int n0, int k0, int t) {
  const int k = t >> 3, n = (t & 7) * 16;
  const float4* g = (const float4*)(src + (size_t)(k0 + k) * ld + (n0 + n));
  float4 f[4]; f[0] = g[0]; f[1] = g[1]; f[2] = g[2]; f[3] = g[3];
  const float* fs = (const float*)f;
#pragma unroll
  for (int j = 0; j < 16; ++j) dst[(n + j) * KSTEP + k] = f2bf(fs[j]);
}

__device__ __forceinline__ void fill_trans_bf16(__bf16* dst, const __bf16* src,
                                                int ld, int n0, int k0, int t) {
  const int k = t >> 3, n = (t & 7) * 16;
  v16bf v = *(const v16bf*)(src + (size_t)(k0 + k) * ld + (n0 + n));
#pragma unroll
  for (int j = 0; j < 16; ++j) dst[(n + j) * KSTEP + k] = v[j];
}

// ---------------- WMMA compute over one K-step ----------------------------
__device__ __forceinline__ v16bf cat8(v8bf lo, v8bf hi) {
  return __builtin_shufflevector(lo, hi, 0, 1, 2, 3, 4, 5, 6, 7,
                                 8, 9, 10, 11, 12, 13, 14, 15);
}

__device__ __forceinline__ void mma_tiles(const __bf16* lA, const __bf16* lB,
                                          v8f acc[2][4], int lane, int wm, int wn) {
  const int m  = lane & 15;
  const int hi = lane >> 4;
  v16bf aF[2], bF[4];
#pragma unroll
  for (int mi = 0; mi < 2; ++mi) {
    // A 16x32 bf16 layout: lanes 0-15 K={0..7,16..23}, lanes 16-31 K={8..15,24..31}
    const __bf16* p = lA + (size_t)(wm * 32 + mi * 16 + m) * KSTEP;
    aF[mi] = cat8(*(const v8bf*)(p + hi * 8), *(const v8bf*)(p + 16 + hi * 8));
  }
#pragma unroll
  for (int ni = 0; ni < 4; ++ni) {
    // B 32x16 bf16 layout: lane selects column, lanes 0-15 K={0..7,16..23}, lanes 16-31 K={8..15,24..31}
    const __bf16* p = lB + (size_t)(wn * 64 + ni * 16 + m) * KSTEP;
    bF[ni] = cat8(*(const v8bf*)(p + hi * 8), *(const v8bf*)(p + 16 + hi * 8));
  }
#pragma unroll
  for (int mi = 0; mi < 2; ++mi)
#pragma unroll
    for (int ni = 0; ni < 4; ++ni)
      acc[mi][ni] = __builtin_amdgcn_wmma_f32_16x16x32_bf16(
          false, aF[mi], false, bF[ni], (short)0, acc[mi][ni], false, false);
}

// ---------------- Kernel 1: QKV projection (3072 x 4096, K=256) -----------
__global__ __launch_bounds__(256)
void k_qkv_proj(const float* __restrict__ wq, const float* __restrict__ x,
                __bf16* __restrict__ qkv) {
  __shared__ __align__(16) __bf16 sA[TM * KSTEP];
  __shared__ __align__(16) __bf16 sB[TN * KSTEP];
  const int t = threadIdx.x, lane = t & 31, wid = t >> 5;
  const int wm = wid & 3, wn = wid >> 2;
  const int m0 = blockIdx.y * TM;   // output channel o
  const int n0 = blockIdx.x * TN;   // spatial n
  v8f acc[2][4] = {};
  for (int k0 = 0; k0 < DIMC; k0 += KSTEP) {
    fill_direct_f32(sA, wq, DIMC, m0, k0, t);
    fill_trans_f32(sB, x, NSP, n0, k0, t);
    if (k0 + KSTEP < DIMC) {
      __builtin_prefetch(x + (size_t)(k0 + KSTEP + (t >> 3)) * NSP + n0 + (t & 7) * 16, 0, 1);
    }
    __syncthreads();
    mma_tiles(sA, sB, acc, lane, wm, wn);
    __syncthreads();
  }
  const int mBase = m0 + wm * 32;
  const int nBase = n0 + wn * 64;
#pragma unroll
  for (int mi = 0; mi < 2; ++mi)
#pragma unroll
    for (int ni = 0; ni < 4; ++ni) {
      const int col = nBase + ni * 16 + (lane & 15);
#pragma unroll
      for (int r = 0; r < 8; ++r) {
        const int row = mBase + mi * 16 + (lane >> 4) * 8 + r;
        const float s = (row < CO) ? 0.0625f : 1.0f;   // q * c^-0.5
        qkv[(size_t)row * NSP + col] = f2bf(acc[mi][ni][r] * s);
      }
    }
}

// ---------------- Kernel 2: logits L[c,d] = q . k (K = 4096) --------------
// Double-buffered pipeline: A tile via Tensor Data Mover (wave 0 issues,
// TENSORcnt), B tile via per-lane GLOBAL_LOAD_ASYNC_TO_LDS (ASYNCcnt).
__global__ __launch_bounds__(256)
void k_logits(const __bf16* __restrict__ qkv, float* __restrict__ attn) {
  __shared__ __align__(16) __bf16 sA[2][TM * KSTEP];
  __shared__ __align__(16) __bf16 sB[2][TN * KSTEP];
  const int t = threadIdx.x, lane = t & 31, wid = t >> 5;
  const int wm = wid & 3, wn = wid >> 2;
  const int h  = blockIdx.z;
  const __bf16* q  = qkv + (size_t)(h * DIMC) * NSP;
  const __bf16* km = qkv + (size_t)(CO + h * DIMC) * NSP;
  const int m0 = blockIdx.y * TM;   // c
  const int n0 = blockIdx.x * TN;   // d
  // per-thread 32B slice of the 128x32 B tile
  const int row = t >> 1, col = (t & 1) * 16;
  const __bf16* gB = km + (size_t)(n0 + row) * NSP + col;
  const int slot = row * KSTEP + col;
  const __bf16* gA = q + (size_t)m0 * NSP;   // block-uniform tile origin

  v8f acc[2][4] = {};
  // prologue: tile for k0 = 0 into buffer 0
  if (wid == 0) tdm_load_2d((unsigned)(size_t)&sA[0][0], gA, NSP, KSTEP, TM);
  async_cp32(&sB[0][slot], gB);
  int cur = 0;
  for (int k0 = 0; k0 < NSP; k0 += KSTEP) {
    wait_async0();                            // this wave's B copies landed
    __builtin_amdgcn_s_wait_tensorcnt(0);     // wave 0's TDM tile landed
    __syncthreads();                          // publish; prev-buffer reads done
    const int nxt = cur ^ 1;
    if (k0 + KSTEP < NSP) {
      if (wid == 0)
        tdm_load_2d((unsigned)(size_t)&sA[nxt][0], gA + (k0 + KSTEP), NSP, KSTEP, TM);
      async_cp32(&sB[nxt][slot], gB + (k0 + KSTEP));
    }
    mma_tiles(sA[cur], sB[cur], acc, lane, wm, wn);
    cur = nxt;
  }
  float* o = attn + (size_t)h * DIMC * DIMC;
  const int mBase = m0 + wm * 32;
  const int nBase = n0 + wn * 64;
#pragma unroll
  for (int mi = 0; mi < 2; ++mi)
#pragma unroll
    for (int ni = 0; ni < 4; ++ni) {
      const int ccol = nBase + ni * 16 + (lane & 15);
#pragma unroll
      for (int r = 0; r < 8; ++r) {
        const int rrow = mBase + mi * 16 + (lane >> 4) * 8 + r;
        o[(size_t)rrow * DIMC + ccol] = acc[mi][ni][r];
      }
    }
}

// ---------------- Kernel 3: row softmax over 256-wide rows ----------------
__global__ __launch_bounds__(256)
void k_softmax(float* __restrict__ attn) {
  const int lane = threadIdx.x & 31, wid = threadIdx.x >> 5;
  float* p = attn + ((size_t)blockIdx.x * 8 + wid) * DIMC;
  float v[8];
  float mx = -3.4e38f;
#pragma unroll
  for (int j = 0; j < 8; ++j) { v[j] = p[j * 32 + lane]; mx = fmaxf(mx, v[j]); }
#pragma unroll
  for (int off = 16; off > 0; off >>= 1) mx = fmaxf(mx, __shfl_xor(mx, off, 32));
  float s = 0.f;
#pragma unroll
  for (int j = 0; j < 8; ++j) { v[j] = __expf(v[j] - mx); s += v[j]; }
#pragma unroll
  for (int off = 16; off > 0; off >>= 1) s += __shfl_xor(s, off, 32);
  const float inv = 1.0f / s;
#pragma unroll
  for (int j = 0; j < 8; ++j) p[j * 32 + lane] = v[j] * inv;
}

// ---------------- Kernel 4: O[c,n] = wts @ v (K = 256) --------------------
__global__ __launch_bounds__(256)
void k_attnv(const float* __restrict__ attn, const __bf16* __restrict__ qkv,
             __bf16* __restrict__ outh) {
  __shared__ __align__(16) __bf16 sA[TM * KSTEP];
  __shared__ __align__(16) __bf16 sB[TN * KSTEP];
  const int t = threadIdx.x, lane = t & 31, wid = t >> 5;
  const int wm = wid & 3, wn = wid >> 2;
  const int h  = blockIdx.z;
  const float*  w = attn + (size_t)h * DIMC * DIMC;
  const __bf16* v = qkv + (size_t)(2 * CO + h * DIMC) * NSP;
  const int m0 = blockIdx.y * TM;   // c
  const int n0 = blockIdx.x * TN;   // n
  v8f acc[2][4] = {};
  for (int k0 = 0; k0 < DIMC; k0 += KSTEP) {
    fill_direct_f32(sA, w, DIMC, m0, k0, t);
    fill_trans_bf16(sB, v, NSP, n0, k0, t);
    __syncthreads();
    mma_tiles(sA, sB, acc, lane, wm, wn);
    __syncthreads();
  }
  const int mBase = m0 + wm * 32;
  const int nBase = n0 + wn * 64;
#pragma unroll
  for (int mi = 0; mi < 2; ++mi)
#pragma unroll
    for (int ni = 0; ni < 4; ++ni) {
      const int col = nBase + ni * 16 + (lane & 15);
#pragma unroll
      for (int r = 0; r < 8; ++r) {
        const int rrow = mBase + mi * 16 + (lane >> 4) * 8 + r;
        outh[(size_t)(h * DIMC + rrow) * NSP + col] = f2bf(acc[mi][ni][r]);
      }
    }
}

// ---------------- Kernel 5: final projection + bias (K = 1024) ------------
__global__ __launch_bounds__(256)
void k_outproj(const float* __restrict__ wo, const __bf16* __restrict__ outh,
               const float* __restrict__ bias, float* __restrict__ out) {
  __shared__ __align__(16) __bf16 sA[TM * KSTEP];
  __shared__ __align__(16) __bf16 sB[TN * KSTEP];
  const int t = threadIdx.x, lane = t & 31, wid = t >> 5;
  const int wm = wid & 3, wn = wid >> 2;
  const int m0 = blockIdx.y * TM;   // o
  const int n0 = blockIdx.x * TN;   // n
  v8f acc[2][4] = {};
  for (int k0 = 0; k0 < CO; k0 += KSTEP) {
    fill_direct_f32(sA, wo, CO, m0, k0, t);
    fill_trans_bf16(sB, outh, NSP, n0, k0, t);
    if (k0 + KSTEP < CO) {
      __builtin_prefetch(outh + (size_t)(k0 + KSTEP + (t >> 3)) * NSP + n0 + (t & 7) * 16, 0, 1);
    }
    __syncthreads();
    mma_tiles(sA, sB, acc, lane, wm, wn);
    __syncthreads();
  }
  const int mBase = m0 + wm * 32;
  const int nBase = n0 + wn * 64;
#pragma unroll
  for (int mi = 0; mi < 2; ++mi)
#pragma unroll
    for (int ni = 0; ni < 4; ++ni) {
      const int col = nBase + ni * 16 + (lane & 15);
#pragma unroll
      for (int r = 0; r < 8; ++r) {
        const int rrow = mBase + mi * 16 + (lane >> 4) * 8 + r;
        out[(size_t)rrow * NSP + col] = acc[mi][ni][r] + bias[rrow];
      }
    }
}

// ---------------- Host launcher -------------------------------------------
extern "C" void kernel_launch(void* const* d_in, const int* in_sizes, int n_in,
                              void* d_out, int out_size, void* d_ws, size_t ws_size,
                              hipStream_t stream) {
  const float* x  = (const float*)d_in[0];   // (8, 256, 64, 64)
  const float* wq = (const float*)d_in[1];   // (3072, 256)
  const float* wo = (const float*)d_in[2];   // (256, 1024)
  const float* bo = (const float*)d_in[3];   // (256,)
  float* out = (float*)d_out;                // (8, 256, 64, 64)

  // Per-batch workspace (reused serially across the batch loop):
  //   qkv  bf16 [3072][4096]  = 25,165,824 B
  //   attn f32  [4][256][256] =  1,048,576 B
  //   outh bf16 [1024][4096]  =  8,388,608 B
  char* ws = (char*)d_ws;
  __bf16* qkv  = (__bf16*)ws;
  float*  attn = (float*)(ws + (size_t)OQKV * NSP * 2);
  __bf16* outh = (__bf16*)(ws + (size_t)OQKV * NSP * 2 + (size_t)HEADS * DIMC * DIMC * 4);

  const dim3 blk(256);
  for (int b = 0; b < NBATCH; ++b) {
    const float* xb = x + (size_t)b * DIMC * NSP;
    float* ob = out + (size_t)b * DIMC * NSP;
    k_qkv_proj<<<dim3(NSP / TN, OQKV / TM), blk, 0, stream>>>(wq, xb, qkv);
    k_logits  <<<dim3(DIMC / TN, DIMC / TM, HEADS), blk, 0, stream>>>(qkv, attn);
    k_softmax <<<dim3(HEADS * DIMC / 8), blk, 0, stream>>>(attn);
    k_attnv   <<<dim3(NSP / TN, DIMC / TM, HEADS), blk, 0, stream>>>(attn, qkv, outh);
    k_outproj <<<dim3(NSP / TN, DIMC / TM), blk, 0, stream>>>(wo, outh, bo, ob);
  }
}
